// Seq2SeqAttention_24120536334943
// MI455X (gfx1250) — matrix operand order repaired and checked by reading, not compiled, not run on hardware
//
#include <hip/hip_runtime.h>
#include <hip/hip_bf16.h>
#include <math.h>

// ---------------------------------------------------------------------------
// Seq2Seq (Bahdanau) attention for MI455X / gfx1250.
//  B=64, S=2048, D=512, U=512.
//  GEMM enc@W1 done in f16 WMMA (v_wmma_f32_16x16x32_f16) with fp32 accum,
//  fused tanh/V-dot epilogue so h_enc [B,S,U] (256MB) is never written.
//  Two streaming passes over encoder_output (2 x 256MB ~ 22us @ 23.3TB/s),
//  ~68.7 GFLOP of matmul in f16 -> roughly balanced compute/memory.
// ---------------------------------------------------------------------------

typedef _Float16 half8_t __attribute__((ext_vector_type(8)));
typedef _Float16 v16h    __attribute__((ext_vector_type(16)));
typedef float    v8f     __attribute__((ext_vector_type(8)));

#define S_LEN   2048
#define DDIM    512
#define UDIM    512
#define BATCH   64

#define M_TILE  128              // seq rows per workgroup
#define N_CHUNK 256              // units per accumulation pass (2 passes)
#define K_CHUNK 32               // K per WMMA step
#define A_PITCH (DDIM + 8)       // halves; +8 pad -> conflict-free ds_load_b128
#define B_PITCH (K_CHUNK + 8)    // halves; +8 pad -> conflict-free

// ---------------- prep: W1 fp32 [D][U] -> f16 transposed [U][D] -------------
__global__ __launch_bounds__(256) void prep_w1(const float* __restrict__ w1,
                                               _Float16* __restrict__ w1t) {
  int idx = blockIdx.x * 256 + threadIdx.x;     // 512*512 / 256 = 1024 blocks
  int u = idx >> 9;
  int d = idx & 511;
  w1t[(u << 9) + d] = (_Float16)w1[(d << 9) + u];
}

// ---------------- prep: hdec[b][u] = dec@W2 + b1 + b2 -----------------------
__global__ __launch_bounds__(256) void prep_hdec(const float* __restrict__ dh,
                                                 const float* __restrict__ w2,
                                                 const float* __restrict__ b1,
                                                 const float* __restrict__ b2,
                                                 float* __restrict__ hdec) {
  int idx = blockIdx.x * 256 + threadIdx.x;     // 64*512 / 256 = 128 blocks
  int b = idx >> 9;
  int u = idx & 511;
  const float* dhb = dh + (b << 9);
  float acc = b1[u] + b2[u];
  #pragma unroll 8
  for (int d = 0; d < DDIM; ++d)
    acc = fmaf(dhb[d], w2[(d << 9) + u], acc);  // coalesced over u
  hdec[idx] = acc;
}

// ---------------- main fused scores kernel (WMMA f16) -----------------------
// grid: 64 batches x 16 m-tiles = 1024 WGs of 256 threads (8 waves).
__global__ __launch_bounds__(256) void att_scores(
    const float*    __restrict__ enc,
    const _Float16* __restrict__ w1t,
    const float*    __restrict__ hdec,
    const float*    __restrict__ vvec,
    float*          __restrict__ scores) {
  extern __shared__ _Float16 smem[];
  _Float16* As = smem;                          // M_TILE x A_PITCH
  _Float16* Bs = smem + M_TILE * A_PITCH;       // N_CHUNK x B_PITCH

  const int bid   = blockIdx.x;
  const int b     = bid >> 4;                   // batch
  const int mtile = bid & 15;
  const int sbase = mtile * M_TILE;
  const int tid   = threadIdx.x;
  const int lane  = tid & 31;
  const int wave  = tid >> 5;
  const int l15   = lane & 15;
  const int kh    = lane >> 4;                  // K-half select per ISA layout

  // ---- stage encoder tile: 128x512 fp32 -> f16 LDS (read once from HBM) ----
  const float* encB = enc + (size_t)(b * S_LEN + sbase) * DDIM;
  for (int i = tid; i < M_TILE * (DDIM / 4); i += 256) {
    int row = i >> 7;                           // / (DDIM/4)
    int c4  = i & 127;
    float4 f = ((const float4*)encB)[(size_t)row * (DDIM / 4) + c4];
    _Float16* dst = As + row * A_PITCH + c4 * 4;
    dst[0] = (_Float16)f.x; dst[1] = (_Float16)f.y;
    dst[2] = (_Float16)f.z; dst[3] = (_Float16)f.w;
  }

  const int mrow = wave * 16 + l15;             // this lane's A row in the tile
  float sp[8];
  #pragma unroll
  for (int v = 0; v < 8; ++v) sp[v] = 0.0f;

  const float* hd = hdec + b * UDIM;

  for (int nc = 0; nc < UDIM; nc += N_CHUNK) {
    v8f acc[16];
    #pragma unroll
    for (int nt = 0; nt < 16; ++nt) acc[nt] = (v8f)0.0f;

    for (int kc = 0; kc < DDIM; kc += K_CHUNK) {
      __syncthreads();                          // prior K-step done with Bs
      // stage W1^T chunk: rows nc..nc+255, k = kc..kc+31 (64B per thread)
      {
        const _Float16* src = w1t + (size_t)(nc + tid) * DDIM + kc;
        _Float16* dst = Bs + tid * B_PITCH;
        half8_t s0 = *(const half8_t*)(src);
        half8_t s1 = *(const half8_t*)(src + 8);
        half8_t s2 = *(const half8_t*)(src + 16);
        half8_t s3 = *(const half8_t*)(src + 24);
        *(half8_t*)(dst)      = s0;
        *(half8_t*)(dst + 8)  = s1;
        *(half8_t*)(dst + 16) = s2;
        *(half8_t*)(dst + 24) = s3;
      }
      __syncthreads();

      // A fragment (16x32 f16, ISA 7.12.2): lane kh selects K-half;
      // VGPR0..3 = K kh*8..kh*8+7, VGPR4..7 = K 16+kh*8..16+kh*8+7.
      const _Float16* ar = As + mrow * A_PITCH + kc + kh * 8;
      half8_t a0 = *(const half8_t*)(ar);
      half8_t a1 = *(const half8_t*)(ar + 16);
      v16h af;
      #pragma unroll
      for (int i = 0; i < 8; ++i) { af[i] = a0[i]; af[8 + i] = a1[i]; }

      #pragma unroll
      for (int nt = 0; nt < 16; ++nt) {
        // B fragment (32x16 f16): lane = column n, K-half kh -> k kh*8..kh*8+7, 16+kh*8..16+kh*8+7
        const _Float16* br = Bs + (nt * 16 + l15) * B_PITCH + kh * 8;
        half8_t b0 = *(const half8_t*)(br);
        half8_t b1f = *(const half8_t*)(br + 16);
        v16h bf;
        #pragma unroll
        for (int i = 0; i < 8; ++i) { bf[i] = b0[i]; bf[8 + i] = b1f[i]; }
        acc[nt] = __builtin_amdgcn_wmma_f32_16x16x32_f16(
            false, af, false, bf, (short)0, acc[nt], false, false);
      }
    }

    // fused epilogue: sp[m] += tanh(acc + hdec[n]) * V[n]
    // C layout: lane holds n = l15, VGPR v holds m = v + 8*kh.
    #pragma unroll
    for (int nt = 0; nt < 16; ++nt) {
      int n = nc + nt * 16 + l15;
      float h = hd[n];
      float vv = vvec[n];
      #pragma unroll
      for (int v = 0; v < 8; ++v)
        sp[v] += tanhf(acc[nt][v] + h) * vv;
    }
  }

  // reduce over the 16 lanes that share each output row m
  #pragma unroll
  for (int v = 0; v < 8; ++v) {
    float x = sp[v];
    #pragma unroll
    for (int off = 8; off >= 1; off >>= 1) x += __shfl_xor(x, off, 16);
    if (l15 == 0) {
      int m = wave * 16 + v + 8 * kh;
      scores[b * S_LEN + sbase + m] = x;
    }
  }
}

// ---------------- softmax over S per batch ----------------------------------
__global__ __launch_bounds__(256) void att_softmax(const float* __restrict__ scores,
                                                   float* __restrict__ attn) {
  __shared__ float red[256];
  int b = blockIdx.x, t = threadIdx.x;
  const float* sb = scores + b * S_LEN;
  float* ab = attn + b * S_LEN;

  float mx = -INFINITY;
  for (int i = t; i < S_LEN; i += 256) mx = fmaxf(mx, sb[i]);
  red[t] = mx; __syncthreads();
  for (int o = 128; o > 0; o >>= 1) {
    if (t < o) red[t] = fmaxf(red[t], red[t + o]);
    __syncthreads();
  }
  mx = red[0]; __syncthreads();

  float sum = 0.0f;
  for (int i = t; i < S_LEN; i += 256) {
    float e = __expf(sb[i] - mx);
    ab[i] = e;
    sum += e;
  }
  red[t] = sum; __syncthreads();
  for (int o = 128; o > 0; o >>= 1) {
    if (t < o) red[t] += red[t + o];
    __syncthreads();
  }
  float inv = 1.0f / red[0];
  for (int i = t; i < S_LEN; i += 256) ab[i] *= inv;
}

// ---------------- zero output -----------------------------------------------
__global__ __launch_bounds__(256) void zero_out(float* __restrict__ out) {
  out[blockIdx.x * 256 + threadIdx.x] = 0.0f;   // 64*512 / 256 = 128 blocks
}

// ---------------- context = sum_s attn * enc (second streaming pass) --------
// grid: 64 batches x 16 seq-chunks of 128; thread t owns d = t and t+256.
__global__ __launch_bounds__(256) void att_context(const float* __restrict__ enc,
                                                   const float* __restrict__ attn,
                                                   float* __restrict__ out) {
  int bid = blockIdx.x;
  int b = bid >> 4, chunk = bid & 15;
  int s0 = chunk * 128;
  int t = threadIdx.x;
  const float* ab = attn + b * S_LEN + s0;
  const float* eb = enc + (size_t)(b * S_LEN + s0) * DDIM;
  float a0 = 0.0f, a1 = 0.0f;
  for (int s = 0; s < 128; ++s) {
    const float* row = eb + (size_t)s * DDIM;
    __builtin_prefetch(row + 4 * DDIM, 0, 1);   // global_prefetch_b8 ahead
    float w = ab[s];
    a0 = fmaf(w, row[t], a0);
    a1 = fmaf(w, row[t + 256], a1);
  }
  atomicAdd(&out[b * DDIM + t], a0);
  atomicAdd(&out[b * DDIM + t + 256], a1);
}

// ---------------------------------------------------------------------------
extern "C" void kernel_launch(void* const* d_in, const int* in_sizes, int n_in,
                              void* d_out, int out_size, void* d_ws, size_t ws_size,
                              hipStream_t stream) {
  const float* enc = (const float*)d_in[0];   // [64,2048,512]
  const float* dh  = (const float*)d_in[1];   // [64,512]
  const float* W1  = (const float*)d_in[2];   // [512,512]
  const float* b1  = (const float*)d_in[3];   // [512]
  const float* W2  = (const float*)d_in[4];   // [512,512]
  const float* b2  = (const float*)d_in[5];   // [512]
  const float* V   = (const float*)d_in[6];   // [512,1]  (bv softmax-invariant)
  float* out = (float*)d_out;                 // [64,512]

  char* ws = (char*)d_ws;
  _Float16* w1t   = (_Float16*)(ws);                                   // 512 KB
  float*    hdec  = (float*)(ws + 512 * 1024);                         // 128 KB
  float*    scores= (float*)(ws + 512 * 1024 + 128 * 1024);            // 512 KB
  float*    attn  = (float*)(ws + 512 * 1024 + 128 * 1024 + 512 * 1024);

  prep_w1  <<<1024, 256, 0, stream>>>(W1, w1t);
  prep_hdec<<<128, 256, 0, stream>>>(dh, W2, b1, b2, hdec);

  size_t lds = (size_t)(M_TILE * A_PITCH + N_CHUNK * B_PITCH) * sizeof(_Float16);
  att_scores<<<BATCH * (S_LEN / M_TILE), 256, lds, stream>>>(enc, w1t, hdec, V, scores);

  att_softmax<<<BATCH, 256, 0, stream>>>(scores, attn);
  zero_out   <<<128, 256, 0, stream>>>(out);
  att_context<<<BATCH * (S_LEN / 128), 256, 0, stream>>>(enc, attn, out);
}
